// GNNEncoder_15195594293396
// MI455X (gfx1250) — matrix operand order repaired and checked by reading, not compiled, not run on hardware
//
#include <hip/hip_runtime.h>
#include <hip/hip_bf16.h>

typedef __attribute__((ext_vector_type(16))) _Float16 v16h;
typedef __attribute__((ext_vector_type(8)))  _Float16 v8h;
typedef __attribute__((ext_vector_type(8)))  float    v8f;

#define DEV __device__ __forceinline__

// Padded LDS row stride for activations: 264 halves = 528 B.
// 528/4 = 132 == 4 (mod 64 banks) -> 16 distinct rows hit disjoint
// 4-bank groups on ds_load_b128 => conflict-free A-fragment gathers.
#define LDS_STRIDE 264

DEV v8f wmma16(v16h a, v16h b, v8f c) {
  return __builtin_amdgcn_wmma_f32_16x16x32_f16(
      false, a, false, b, (short)0, c, false, false);
}

DEV v16h cat8(v8h lo, v8h hi) {
  return __builtin_shufflevector(lo, hi,
      0,1,2,3,4,5,6,7,8,9,10,11,12,13,14,15);
}

// A fragment (16x32 f16) from padded LDS activations.
// Lane: row m = l16; halves[0..7] = K kb+h16*8+0..7, halves[8..15] = +16.
DEV v16h load_a_lds(const _Float16* s, int row, int kb, int h16) {
  const _Float16* p = s + row * LDS_STRIDE + kb + h16 * 8;
  return cat8(*(const v8h*)(p), *(const v8h*)(p + 16));
}

// B fragment (32x16 f16) from an XOR-swizzled 32-K LDS weight slice.
// Slice layout: per column, four 8-half chunks stored at chunk index
// (q ^ ((col>>2)&3)) -> all 32 lanes hit disjoint 4-bank groups.
DEV v16h load_b_sw(const _Float16* s, int col, int h16) {
  const v8h* p = (const v8h*)s;
  const int g = (col >> 2) & 3;
  return cat8(p[col * 4 + (h16 ^ g)],
              p[col * 4 + ((h16 + 2) ^ g)]);
}

// Cooperatively stage one 32-K slice (all 256 cols) of pre-transposed
// weights Wt[out][in] into a 16 KB swizzled LDS buffer. 256 threads.
DEV void stage_w(const _Float16* __restrict__ W, int kb, _Float16* dst,
                 int t) {
  uint4* d = (uint4*)dst;
#pragma unroll
  for (int i = 0; i < 4; i++) {
    const int u = t + i * 256;            // 1024 x 16B
    const int col = u >> 2, q = u & 3;
    d[col * 4 + (q ^ ((col >> 2) & 3))] =
        *(const uint4*)(W + (size_t)col * 256 + kb + q * 8);
  }
}

DEV float relu_f(float v) { return v > 0.0f ? v : 0.0f; }

// ---- weight pack: dst[n*K + k] = (f16) src[k*N + n]
__global__ void k_pack_wt(const float* __restrict__ src,
                          _Float16* __restrict__ dst, int K, int N) {
  int id = blockIdx.x * blockDim.x + threadIdx.x;
  if (id < K * N) {
    int n = id / K, k = id - n * K;
    dst[id] = (_Float16)src[k * N + n];
  }
}

__global__ void k_cvt_f16(const float* __restrict__ src,
                          _Float16* __restrict__ dst, int n) {
  int id = blockIdx.x * blockDim.x + threadIdx.x;
  if (id < n) dst[id] = (_Float16)src[id];
}

// ---- fused 2-layer MLP: Y = relu(relu(X@W1+b1)@W2+b2), X:[R,256] f16.
// WG = 64 rows x 256 cols, 8 waves; weights double-buffered through LDS.
__global__ __launch_bounds__(256) void k_mlp2(
    const _Float16* __restrict__ X,
    const _Float16* __restrict__ W1t, const float* __restrict__ b1,
    const _Float16* __restrict__ W2t, const float* __restrict__ b2,
    _Float16* __restrict__ Yh, float* __restrict__ Yf) {
  __shared__ _Float16 sX[64 * LDS_STRIDE];   // ~33 KB
  __shared__ _Float16 sZ[64 * LDS_STRIDE];   // ~33 KB
  __shared__ _Float16 sW[2][256 * 32];       // 2 x 16 KB
  const int t = threadIdx.x;
  const int wave = t >> 5, lane = t & 31;
  const int l16 = lane & 15, h16 = lane >> 4;
  const int rowBase = blockIdx.x * 64;

  {  // stage X (padded rows)
#pragma unroll
    for (int i = 0; i < 8; i++) {
      const int u = t + i * 256;            // 2048 x 16B
      const int row = u >> 4, seg = u & 15;
      *(uint4*)(sX + row * LDS_STRIDE + seg * 16) =
          *(const uint4*)(X + (size_t)(rowBase + row) * 256 + seg * 16);
    }
  }
  stage_w(W1t, 0, sW[0], t);
  __syncthreads();

  const int rt = wave & 3;
  const int nb0 = (wave >> 2) * 128;
  const int arow = rt * 16;

  // ---- layer 1 (double-buffered weight slices)
  v8f c[8] = {};
#pragma unroll
  for (int k = 0; k < 8; k++) {
    if (k < 7) stage_w(W1t, (k + 1) * 32, sW[(k + 1) & 1], t);
    v16h a = load_a_lds(sX, arow + l16, k * 32, h16);
    const _Float16* wb = sW[k & 1];
#pragma unroll
    for (int j = 0; j < 8; j++)
      c[j] = wmma16(a, load_b_sw(wb, nb0 + j * 16 + l16, h16), c[j]);
    __syncthreads();
  }
#pragma unroll
  for (int j = 0; j < 8; j++) {
    const int col = nb0 + j * 16 + l16;
    const float bb = b1[col];
#pragma unroll
    for (int v = 0; v < 8; v++)
      sZ[(arow + v + 8 * h16) * LDS_STRIDE + col] =
          (_Float16)relu_f(c[j][v] + bb);
  }
  stage_w(W2t, 0, sW[0], t);
  __syncthreads();

  // ---- layer 2
  v8f d[8] = {};
#pragma unroll
  for (int k = 0; k < 8; k++) {
    if (k < 7) stage_w(W2t, (k + 1) * 32, sW[(k + 1) & 1], t);
    v16h a = load_a_lds(sZ, arow + l16, k * 32, h16);
    const _Float16* wb = sW[k & 1];
#pragma unroll
    for (int j = 0; j < 8; j++)
      d[j] = wmma16(a, load_b_sw(wb, nb0 + j * 16 + l16, h16), d[j]);
    __syncthreads();
  }
#pragma unroll
  for (int j = 0; j < 8; j++) {
    const int col = nb0 + j * 16 + l16;
    const float bb = b2[col];
#pragma unroll
    for (int v = 0; v < 8; v++) {
      float val = relu_f(d[j][v] + bb);
      const size_t gr = (size_t)(rowBase + arow + v + 8 * h16);
      if (Yh) Yh[gr * 256 + col] = (_Float16)val;
      if (Yf) Yf[gr * 256 + col] = val;
    }
  }
}

// ---- fused edge pass: 1 WG = 1 batch. gather -> 2-layer edge MLP ->
// scaled scatter-sum into LDS accumulator -> agg f16.
__global__ __launch_bounds__(256) void k_edge(
    const _Float16* __restrict__ H, const int* __restrict__ sendIdx,
    const int* __restrict__ recIdx, int E,
    const _Float16* __restrict__ W1aT, const _Float16* __restrict__ W1bT,
    const float* __restrict__ b1, const _Float16* __restrict__ W2T,
    const float* __restrict__ b2, _Float16* __restrict__ Agg,
    float invdeg) {
  __shared__ _Float16 sH[32 * LDS_STRIDE];   // ~16.5 KB
  __shared__ float sAgg[32 * 256];           // 32 KB
  __shared__ _Float16 sZ[64 * LDS_STRIDE];   // ~33 KB
  __shared__ _Float16 sW[2][2 * 256 * 32];   // 2 x 32 KB (two matrices)
  const int t = threadIdx.x;
  const int wave = t >> 5, lane = t & 31;
  const int l16 = lane & 15, h16 = lane >> 4;
  const int b = blockIdx.x;

  {  // stage h[b] into padded rows
#pragma unroll
    for (int i = 0; i < 2; i++) {
      const int u = t + i * 256;             // 512 x 16B
      const int row = u >> 4, seg = u & 15;
      *(uint4*)(sH + row * LDS_STRIDE + seg * 16) =
          *(const uint4*)(H + (size_t)b * 8192 + row * 256 + seg * 16);
    }
  }
  {  // zero accumulator
    float4* s4 = (float4*)sAgg;
#pragma unroll
    for (int i = 0; i < 8; i++) s4[t + i * 256] = make_float4(0, 0, 0, 0);
  }

  const int p = wave >> 1;           // row-tile slot 0..3 (16 edges each)
  const int nb0 = (wave & 1) * 128;  // column half
  const int zrow = p * 16;
  const int numTiles = E >> 4;       // 62 for E=992
  const int iters = (numTiles + 3) >> 2;

  for (int it = 0; it < iters; ++it) {
    const int rt = it * 4 + p;
    const bool act = rt < numTiles;
    const int eb = rt * 16;
    int sr = 0, rr = 0;
    if (act) {
      const int e = eb + l16;
      sr = sendIdx[e];
      rr = recIdx[e];
    }

    // ---- layer 1: z = relu(h[send]@W1a + h[rec]@W1b + b1)
    stage_w(W1aT, 0, sW[0], t);
    stage_w(W1bT, 0, sW[0] + 8192, t);
    __syncthreads();
    v8f c[8] = {};
#pragma unroll
    for (int k = 0; k < 8; k++) {
      if (k < 7) {
        stage_w(W1aT, (k + 1) * 32, sW[(k + 1) & 1], t);
        stage_w(W1bT, (k + 1) * 32, sW[(k + 1) & 1] + 8192, t);
      }
      if (act) {
        v16h as = load_a_lds(sH, sr, k * 32, h16);
        v16h ar = load_a_lds(sH, rr, k * 32, h16);
        const _Float16* wa = sW[k & 1];
        const _Float16* wbm = sW[k & 1] + 8192;
#pragma unroll
        for (int j = 0; j < 8; j++) {
          const int n = nb0 + j * 16 + l16;
          c[j] = wmma16(as, load_b_sw(wa, n, h16), c[j]);
          c[j] = wmma16(ar, load_b_sw(wbm, n, h16), c[j]);
        }
      }
      __syncthreads();
    }
    if (act) {
#pragma unroll
      for (int j = 0; j < 8; j++) {
        const int col = nb0 + j * 16 + l16;
        const float bb = b1[col];
#pragma unroll
        for (int v = 0; v < 8; v++)
          sZ[(zrow + v + 8 * h16) * LDS_STRIDE + col] =
              (_Float16)relu_f(c[j][v] + bb);
      }
    }

    // ---- layer 2 + scaled scatter into sAgg
    stage_w(W2T, 0, sW[0], t);
    __syncthreads();  // covers sZ writes + slice 0
    v8f d[8] = {};
#pragma unroll
    for (int k = 0; k < 8; k++) {
      if (k < 7) stage_w(W2T, (k + 1) * 32, sW[(k + 1) & 1], t);
      if (act) {
        v16h a = load_a_lds(sZ, zrow + l16, k * 32, h16);
        const _Float16* wb = sW[k & 1];
#pragma unroll
        for (int j = 0; j < 8; j++)
          d[j] = wmma16(a, load_b_sw(wb, nb0 + j * 16 + l16, h16), d[j]);
      }
      __syncthreads();
    }
    if (act) {
      int rr2[8];
#pragma unroll
      for (int v = 0; v < 8; v++) rr2[v] = recIdx[eb + v + 8 * h16];
#pragma unroll
      for (int j = 0; j < 8; j++) {
        const int col = nb0 + j * 16 + l16;
        const float bb = b2[col];
#pragma unroll
        for (int v = 0; v < 8; v++)
          atomicAdd(&sAgg[rr2[v] * 256 + col],
                    relu_f(d[j][v] + bb) * invdeg);
      }
    }
  }
  __syncthreads();

  // write agg (already mean) as f16
  _Float16* dst = Agg + (size_t)b * 8192;
#pragma unroll
  for (int i = 0; i < 32; i++) {
    const int idx = t + i * 256;
    dst[idx] = (_Float16)sAgg[idx];
  }
}

extern "C" void kernel_launch(void* const* d_in, const int* in_sizes, int n_in,
                              void* d_out, int out_size, void* d_ws,
                              size_t ws_size, hipStream_t stream) {
  const float* x     = (const float*)d_in[0];
  const int*   rec   = (const int*)d_in[1];
  const int*   snd   = (const int*)d_in[2];
  const float* w_in1 = (const float*)d_in[3];
  const float* b_in1 = (const float*)d_in[4];
  const float* w_in2 = (const float*)d_in[5];
  const float* b_in2 = (const float*)d_in[6];
  const float* w_e1  = (const float*)d_in[7];
  const float* b_e1  = (const float*)d_in[8];
  const float* w_e2  = (const float*)d_in[9];
  const float* b_e2  = (const float*)d_in[10];
  const float* w_n1  = (const float*)d_in[11];
  const float* b_n1  = (const float*)d_in[12];
  const float* w_n2  = (const float*)d_in[13];
  const float* b_n2  = (const float*)d_in[14];

  const int E = in_sizes[1];       // 992
  const int total = in_sizes[0];   // B*N*256 = 2097152
  const int R = total / 256;       // 8192 rows
  const int Bsz = R / 32;          // 256 batches
  float* out = (float*)d_out;

  char* ws = (char*)d_ws;
  const size_t MAT = (size_t)256 * 256 * sizeof(_Float16);  // 128 KB
  _Float16* Win1T = (_Float16*)(ws + 0 * MAT);
  _Float16* Win2T = (_Float16*)(ws + 1 * MAT);
  _Float16* We1aT = (_Float16*)(ws + 2 * MAT);
  _Float16* We1bT = (_Float16*)(ws + 3 * MAT);
  _Float16* We2T  = (_Float16*)(ws + 4 * MAT);
  _Float16* Wn1T  = (_Float16*)(ws + 5 * MAT);
  _Float16* Wn2T  = (_Float16*)(ws + 6 * MAT);
  _Float16* xh    = (_Float16*)(ws + 7 * MAT);
  _Float16* h16   = xh + (size_t)total;
  _Float16* agg16 = h16 + (size_t)total;

  dim3 pb(256), pg((256 * 256 + 255) / 256);
  k_pack_wt<<<pg, pb, 0, stream>>>(w_in1, Win1T, 256, 256);
  k_pack_wt<<<pg, pb, 0, stream>>>(w_in2, Win2T, 256, 256);
  k_pack_wt<<<pg, pb, 0, stream>>>(w_e1,             We1aT, 256, 256);
  k_pack_wt<<<pg, pb, 0, stream>>>(w_e1 + 256 * 256, We1bT, 256, 256);
  k_pack_wt<<<pg, pb, 0, stream>>>(w_e2, We2T, 256, 256);
  k_pack_wt<<<pg, pb, 0, stream>>>(w_n1, Wn1T, 256, 256);
  k_pack_wt<<<pg, pb, 0, stream>>>(w_n2, Wn2T, 256, 256);
  k_cvt_f16<<<(total + 255) / 256, 256, 0, stream>>>(x, xh, total);

  const float invdeg = 1.0f / 31.0f;  // fully-connected: deg = N-1
  const int mlpGrid = R / 64;

  k_mlp2<<<mlpGrid, 256, 0, stream>>>(xh, Win1T, b_in1, Win2T, b_in2,
                                      h16, nullptr);
  k_edge<<<Bsz, 256, 0, stream>>>(h16, snd, rec, E, We1aT, We1bT, b_e1,
                                  We2T, b_e2, agg16, invdeg);
  k_mlp2<<<mlpGrid, 256, 0, stream>>>(agg16, Wn1T, b_n1, Wn2T, b_n2,
                                      h16, nullptr);
  k_edge<<<Bsz, 256, 0, stream>>>(h16, snd, rec, E, We1aT, We1bT, b_e1,
                                  We2T, b_e2, agg16, invdeg);
  k_mlp2<<<mlpGrid, 256, 0, stream>>>(agg16, Wn1T, b_n1, Wn2T, b_n2,
                                      nullptr, out);
}
